// Gaussian2D_75033078661124
// MI455X (gfx1250) — compile-verified, matrix operand order checked by reading, not run on hardware
//
#include <hip/hip_runtime.h>
#include <hip/hip_bf16.h>

typedef __attribute__((ext_vector_type(16))) _Float16 v16h;
typedef __attribute__((ext_vector_type(8)))  float    v8f;

#define NG   1024
#define IMH  768
#define IMW  768

// workspace layout (floats)
#define WS_IA   0          // d/det
#define WS_BC   (NG)       // -(b+c)/det
#define WS_ID   (2*NG)     // a/det
#define WS_MX   (3*NG)
#define WS_MY   (4*NG)
#define WS_RGB  (5*NG)     // NG * float4 {wr, wg, wb, 0}

// Phase 1: per-gaussian derived parameters (trivial, 1024 elements).
__global__ void gauss_pre(const float* __restrict__ means,
                          const float* __restrict__ cov,
                          const float* __restrict__ colors,
                          float* __restrict__ ws) {
  int g = blockIdx.x * blockDim.x + threadIdx.x;
  if (g >= NG) return;
  float a = cov[g*4+0], b = cov[g*4+1], c = cov[g*4+2], d = cov[g*4+3];
  float inv = 1.0f / (a*d - b*c);
  ws[WS_IA + g] = d * inv;
  ws[WS_BC + g] = -(b + c) * inv;
  ws[WS_ID + g] = a * inv;
  ws[WS_MX + g] = means[g*2+0];
  ws[WS_MY + g] = means[g*2+1];
  float alpha = colors[g*4+3];
  float4 r;
  r.x = alpha / (1.0f + __expf(-colors[g*4+0]));
  r.y = alpha / (1.0f + __expf(-colors[g*4+1]));
  r.z = alpha / (1.0f + __expf(-colors[g*4+2]));
  r.w = 0.0f;                       // dead WMMA columns read exact zero
  ((float4*)(ws + WS_RGB))[g] = r;
}

// Phase 2: one block = one image row h, 128 consecutive pixels (8 waves x 16).
// Each wave: C(16 pixels x 16 channels, f32) accumulated over K=1024 gaussians
// via 32 x v_wmma_f32_16x16x32_f16; exp evaluation co-executes with WMMA
// (16-bit XDL WMMA is TRANS-tracked on CDNA5).
__global__ __launch_bounds__(256) void gauss_splat(const float* __restrict__ ws,
                                                   float* __restrict__ out) {
  __shared__ float4 sP[NG];      // {A_h, B_h, id, my} per gaussian, 16 KB
  __shared__ float4 sRgb[NG];    // {wr, wg, wb, 0} per gaussian,   16 KB

  const int h     = blockIdx.x / (IMW / 128);
  const int wtile = blockIdx.x % (IMW / 128);
  const float xh  = (float)h * (1.0f / (IMH - 1));

  // Row-dependent per-gaussian precompute: dx depends only on h.
  for (int g = threadIdx.x; g < NG; g += 256) {
    float dx = xh - ws[WS_MX + g];
    float4 p;
    p.x = ws[WS_IA + g] * dx * dx;   // A_h = ia*dx^2
    p.y = ws[WS_BC + g] * dx;        // B_h = (ib+ic)*dx
    p.z = ws[WS_ID + g];             // id
    p.w = ws[WS_MY + g];             // my
    sP[g]   = p;
    sRgb[g] = ((const float4*)(ws + WS_RGB))[g];
  }
  __syncthreads();

  const int wave = threadIdx.x >> 5;
  const int lane = threadIdx.x & 31;
  const int m    = lane & 15;        // A-row: pixel within wave tile; B-col: channel
  const int hi   = lane >> 4;        // lane-half selects K sub-range
  const int w0   = wtile * 128 + wave * 16;
  const float yw = (float)(w0 + m) * (1.0f / (IMW - 1));
  const int cidx = m & 3;            // channel slot within float4 (wrapped; masked below)

  v8f acc = {};
  for (int chunk = 0; chunk < 32; ++chunk) {
    const int gbase = chunk * 32 + hi * 8;
    v16h av, bv;
#pragma unroll
    for (int j = 0; j < 16; ++j) {
      // K mapping for 16-bit operands (wave32):
      //   j<8  -> K = hi*8 + j ; j>=8 -> K = 16 + hi*8 + (j-8)
      const int g = gbase + j + ((j >= 8) ? 8 : 0);
      float4 p = sP[g];                          // one ds_load_b128, lane-half broadcast
      float dy = yw - p.w;
      float q  = p.x + dy * (p.y + dy * p.z);
      // exp(-q/2) = 2^(-0.72134752*q): raw v_exp_f32, no ldexp range fixup
      av[j] = (_Float16)__builtin_amdgcn_exp2f(q * -0.72134752f);
      float rv = ((const float*)sRgb)[(g << 2) + cidx];
      bv[j] = (_Float16)((m < 4) ? rv : 0.0f);   // channels 3..15 are zero columns
    }
    // D = A(16x32 f16 vals) x B(32x16 f16 wrgb) + C ; EXEC all-ones here.
    acc = __builtin_amdgcn_wmma_f32_16x16x32_f16(
        /*neg_a=*/false, av, /*neg_b=*/false, bv,
        /*c_mod=*/(short)0, acc, /*reuse_a=*/false, /*reuse_b=*/false);
  }

  // C/D layout: lane<16 -> rows M=0..7 in acc[0..7]; lane>=16 -> rows M=8..15.
  if (m < 3) {
#pragma unroll
    for (int v = 0; v < 8; ++v) {
      int px = w0 + v + hi * 8;
      out[((size_t)h * IMW + px) * 3 + m] = acc[v];
    }
  }
}

extern "C" void kernel_launch(void* const* d_in, const int* in_sizes, int n_in,
                              void* d_out, int out_size, void* d_ws, size_t ws_size,
                              hipStream_t stream) {
  const float* means  = (const float*)d_in[0];   // (N,2,1) f32
  const float* cov    = (const float*)d_in[1];   // (N,2,2) f32
  const float* colors = (const float*)d_in[2];   // (N,4)   f32
  float* ws  = (float*)d_ws;
  float* out = (float*)d_out;                    // (H,W,3) f32

  gauss_pre<<<(NG + 255) / 256, 256, 0, stream>>>(means, cov, colors, ws);
  gauss_splat<<<IMH * (IMW / 128), 256, 0, stream>>>(ws, out);
}
